// ChebychevInput_25847113187936
// MI455X (gfx1250) — matrix operand order repaired and checked by reading, not compiled, not run on hardware
//
#include <hip/hip_runtime.h>

// ============================================================================
// Chebyshev-feature GEMM for MI455X (gfx1250, wave32, WMMA).
//
//   out[b,o,s] = sum_{i,p} T_p(x[b,i,s]) * (WMAG * coeff[o,i,p])
//
// GEMM view: M = B*S = 262144, N = O = 256, K = 3*257 = 771 (padded 3*288=864).
// ~103 GFLOP vs ~272 MB HBM traffic -> compute bound -> f16 WMMA, f32 accum.
//
// Kernel 1 packs scaled f16 weights into d_ws in exact per-lane A-fragment
// order for v_wmma_f32_16x16x32_f16 (contiguous k = 16h + e, same order as B).
// Kernel 2 generates Chebyshev B-fragments in-register via the recurrence
// T_p = 2x*T_{p-1} - T_{p-2}. To avoid the duplicated recurrence across
// half-waves (B columns repeat for lanes n and n+16), each half-wave computes
// only its 2 of the 4 s-tiles and the halves exchange fragment-halves with
// ds_swizzle_b32 SWAPX16 (bit-exact data movement on the DS-permute pipe).
// Per k-chunk: ~145 VALU + 16 DS ops feeding 32 WMMAs.
// ============================================================================

typedef __attribute__((ext_vector_type(16))) _Float16 v16h;
typedef __attribute__((ext_vector_type(8)))  float    v8f;

#define IN_DIM   3
#define OUT_DIM  256
#define PP1      257          // poly_degree + 1
#define KP_I     288          // padded K per input dim (9 chunks of 32)
#define NCH_I    9            // k-chunks per input dim
#define KCH      (IN_DIM * NCH_I)   // 27 total k-chunks (K_pad = 864)
#define NSAMP    131072
#define BATCH    2
#define NOTILE   16           // 256 outputs / 16

// ---------------------------------------------------------------------------
// Kernel 1: pack weights into WMMA A-fragment order, f16, scaled, K-padded.
// ws layout: v16h frag[kc][ot][lane], kc<27, ot<16, lane<32  (432 KB)
// A-frag element e of lane (h = lane/16, m = lane%16):
//   row M = m  (-> output o = ot*16 + m)
//   K    = kc*32 + e + 16*h
// K decomposes as i = K/288, p = K%288; p > 256 is zero padding.
// ---------------------------------------------------------------------------
__global__ void cheb_pack_weights(const float* __restrict__ coeff,
                                  _Float16* __restrict__ wpk) {
  int tid = blockIdx.x * blockDim.x + threadIdx.x;   // (kc*16 + ot)*32 + lane
  if (tid >= KCH * NOTILE * 32) return;
  const int lane = tid & 31;
  const int ot   = (tid >> 5) & 15;
  const int kc   = tid >> 9;
  const int h    = lane >> 4;
  const int o    = ot * 16 + (lane & 15);
  const float wmag = sqrtf(6.0f / (float)(IN_DIM * PP1));

  v16h frag;
#pragma unroll
  for (int e = 0; e < 16; ++e) {
    const int koff = e + 16 * h;
    const int k = kc * 32 + koff;
    const int i = k / KP_I;
    const int p = k % KP_I;
    float v = 0.0f;
    if (p < PP1) v = wmag * coeff[(o * IN_DIM + i) * PP1 + p];
    frag[e] = (_Float16)v;
  }
  ((v16h*)wpk)[tid] = frag;
}

// Swap a v16h between lane pairs (n, n+16): ds_swizzle group-of-32,
// xor_mask=0x10, or=0, and=0x1f  -> pattern 0x401f (SWAPX16).
static __device__ __forceinline__ v16h swapx16(v16h v) {
  union { v16h h; int i[8]; } u;
  u.h = v;
#pragma unroll
  for (int d = 0; d < 8; ++d)
    u.i[d] = __builtin_amdgcn_ds_swizzle(u.i[d], 0x401f);
  return u.h;
}

// ---------------------------------------------------------------------------
// Kernel 2: main WMMA GEMM.
// Block = 256 threads = 8 waves. Waves 0-3: o-half 0, waves 4-7: o-half 1.
// Each wave: 64 samples (4 s-tiles of 16) x 128 outputs (8 o-tiles of 16).
// Block covers 256 samples x all 256 outputs. Grid = B*S/256 = 1024 blocks.
//
// B-fragment (T-matrix, 32x16 f16) layout (contiguous k = 16h + e, same order as A):
//   lane n (0-15) holds column N=n, K = 0..15 contiguous in element order;
//   lane n+16 holds column N=n, K = 16..31.
// Low half-wave lanes run the recurrence for s-tiles {0,1}; high half-wave
// lanes for s-tiles {2,3}; SWAPX16 delivers the missing fragment-halves.
// ---------------------------------------------------------------------------
__global__ __launch_bounds__(256)
void cheb_wmma_gemm(const float* __restrict__ x,
                    const _Float16* __restrict__ wpk,
                    float* __restrict__ out) {
  const int lane = threadIdx.x & 31;
  const int wave = threadIdx.x >> 5;
  const int n    = lane & 15;          // sample offset within a tile (B col)
  const int h    = lane >> 4;          // half-wave id
  const bool hilane = (h != 0);

  const int o_half = wave >> 2;        // 0 -> outputs 0..127, 1 -> 128..255
  const long g  = (long)blockIdx.x * 256 + (long)(wave & 3) * 64;  // flat (b,s)
  const int  b  = (int)(g / NSAMP);    // 512 blocks per batch, never straddles
  const int  s0 = (int)(g % NSAMP);

  const v16h* __restrict__ wv = (const v16h*)wpk;

  v8f acc[4][8] = {};                  // [s_tile][o_tile] f32 accumulators

  for (int i = 0; i < IN_DIM; ++i) {
    // This lane computes recurrences for s-tiles (2h + j), j = 0,1.
    float x2[2], tc[2], tp[2];
#pragma unroll
    for (int j = 0; j < 2; ++j) {
      const int st = 2 * h + j;
      const float xv = x[((long)b * IN_DIM + i) * NSAMP + s0 + st * 16 + n];
      x2[j] = xv + xv;
      tc[j] = 1.0f;   // T_0
      tp[j] = xv;     // seed so first advance yields T_1 = 2x*1 - x = x
    }

    for (int pc = 0; pc < NCH_I; ++pc) {
      // 32 recurrence steps for 2 columns; split captures into lo/hi halves.
      v16h mlo[2], mhi[2];
#pragma unroll
      for (int t = 0; t < 32; ++t) {
#pragma unroll
        for (int j = 0; j < 2; ++j) {
          const _Float16 hv = (_Float16)tc[j];
          if (t < 16) mlo[j][t] = hv;
          else        mhi[j][t - 16] = hv;
          const float tn = __builtin_fmaf(x2[j], tc[j], -tp[j]);
          tp[j] = tc[j];
          tc[j] = tn;
        }
      }

      // Cross-half exchange: lo lanes send hi-half of tiles {0,1} and receive
      // lo-half of tiles {2,3}; hi lanes vice-versa. One SWAPX16 per j.
      v16h bfrag[4];
#pragma unroll
      for (int j = 0; j < 2; ++j) {
        const v16h xch = swapx16(hilane ? mlo[j] : mhi[j]);
        bfrag[0 + j] = hilane ? xch    : mlo[j];  // tile 2h'+j with h'=0
        bfrag[2 + j] = hilane ? mhi[j] : xch;     // tile 2h'+j with h'=1
      }

      const int kc = i * NCH_I + pc;
#pragma unroll
      for (int ot = 0; ot < 8; ++ot) {
        // W A-fragment: 32 B/lane from L2-resident packed buffer.
        const v16h afrag = wv[((kc * NOTILE) + (o_half * 8 + ot)) * 32 + lane];
#pragma unroll
        for (int st = 0; st < 4; ++st) {
          acc[st][ot] = __builtin_amdgcn_wmma_f32_16x16x32_f16(
              /*neg_a=*/false, afrag, /*neg_b=*/false, bfrag[st],
              /*c_mod=*/(short)0, acc[st][ot],
              /*reuse_a=*/false, /*reuse_b=*/false);
        }
      }
    }
  }

  // Store. D layout: lane n = column (sample), VGPR v -> row M = v + 8*h.
  const int hoff = hilane ? 8 : 0;
#pragma unroll
  for (int ot = 0; ot < 8; ++ot) {
    const int obase = o_half * 128 + ot * 16 + hoff;
#pragma unroll
    for (int st = 0; st < 4; ++st) {
#pragma unroll
      for (int v = 0; v < 8; ++v) {
        out[((size_t)b * OUT_DIM + obase + v) * NSAMP + s0 + st * 16 + n] =
            acc[st][ot][v];
      }
    }
  }
}

// ---------------------------------------------------------------------------
extern "C" void kernel_launch(void* const* d_in, const int* in_sizes, int n_in,
                              void* d_out, int out_size, void* d_ws, size_t ws_size,
                              hipStream_t stream) {
  const float* x     = (const float*)d_in[0];  // (2, 3, 131072) f32
  const float* coeff = (const float*)d_in[1];  // (256, 3, 257) f32
  float* out         = (float*)d_out;          // (2, 256, 131072) f32
  _Float16* wpk      = (_Float16*)d_ws;        // 27*16*32*16 f16 = 432 KB

  (void)in_sizes; (void)n_in; (void)out_size; (void)ws_size;

  // Kernel 1: pack weights (13824 fragment-owning threads).
  {
    const int total = KCH * NOTILE * 32;
    const int blk = 256;
    cheb_pack_weights<<<(total + blk - 1) / blk, blk, 0, stream>>>(coeff, wpk);
  }
  // Kernel 2: WMMA GEMM. 1024 blocks x 256 threads (8 waves each).
  {
    const int blocks = (BATCH * NSAMP) / 256;   // 1024
    cheb_wmma_gemm<<<blocks, 256, 0, stream>>>(x, wpk, out);
  }
}
